// SAE_32143535243594
// MI455X (gfx1250) — matrix operand order repaired and checked by reading, not compiled, not run on hardware
//
#include <hip/hip_runtime.h>
#include <hip/hip_bf16.h>

// ---------------- problem constants ----------------
#define D_MODEL 2048
#define N_FEAT  32768
#define BATCH   4096
#define TOPK    32

// ---------------- tiling ----------------
#define BM 128
#define BN 64
#define BK 32
#define BS 40                      // LDS row stride (elems) for Bt tiles
#define NSPLIT 16                  // feature-dim split for parallelism
#define NCHUNK (N_FEAT / NSPLIT)   // 2048
#define NTILES (NCHUNK / BN)       // 32
#define KTILES (D_MODEL / BK)      // 64
#define NCAND  (NSPLIT * TOPK)     // 512 candidates per row

typedef __attribute__((ext_vector_type(16))) __bf16 v16bf;
typedef __attribute__((ext_vector_type(8)))  __bf16 v8bf;
typedef __attribute__((ext_vector_type(4)))  __bf16 v4bf;
typedef __attribute__((ext_vector_type(8)))  float  v8f;

#define SQRT_D 45.25483399593904f  // sqrt(2048)

__device__ __forceinline__ v16bf frag_cat(const __bf16* p0, const __bf16* p1) {
  v8bf a = *(const v8bf*)p0;   // 16B -> b128 load (global or LDS)
  v8bf b = *(const v8bf*)p1;
  return __builtin_shufflevector(a, b, 0,1,2,3,4,5,6,7,8,9,10,11,12,13,14,15);
}

__device__ __forceinline__ v8f wmma_bf16(v16bf a, v16bf b, v8f c) {
  return __builtin_amdgcn_wmma_f32_16x16x32_bf16(false, a, false, b, (short)0, c, false, false);
}

__device__ __forceinline__ void split_hl(float f, __bf16& hi, __bf16& lo) {
  __bf16 h = (__bf16)f;            // RNE
  hi = h;
  lo = (__bf16)(f - (float)h);     // residual
}

// ============================================================
// Kernel 0: one-time split of x_normed - b_pre into bf16 hi/lo.
// Result (32 MB) is L2-resident for the whole encode GEMM.
// ============================================================
__global__ __launch_bounds__(256) void sae_split_x(
    const float* __restrict__ x, const float* __restrict__ b_pre,
    const float* __restrict__ avg_norm,
    __bf16* __restrict__ x_hi, __bf16* __restrict__ x_lo)
{
  const size_t base = ((size_t)blockIdx.x * 256 + threadIdx.x) * 4;
  const float scale = SQRT_D / avg_norm[0];
  const float4 xv = *(const float4*)(x + base);
  const float4 bp = *(const float4*)(b_pre + (base & (D_MODEL - 1)));
  float f[4] = { xv.x * scale - bp.x, xv.y * scale - bp.y,
                 xv.z * scale - bp.z, xv.w * scale - bp.w };
  v4bf h, l;
#pragma unroll
  for (int j = 0; j < 4; j++) { __bf16 hh, ll; split_hl(f[j], hh, ll); h[j] = hh; l[j] = ll; }
  *(v4bf*)(x_hi + base) = h;   // b64 stores
  *(v4bf*)(x_lo + base) = l;
}

// ============================================================
// Kernel 1: encode GEMM (bf16 hi/lo split WMMA; A frags straight
// from global/L2, B staged+transposed in LDS) + running exact
// per-row top-32 over this block's feature chunk.
// ============================================================
__global__ __launch_bounds__(256) void sae_encode_topk(
    const __bf16* __restrict__ x_hi, const __bf16* __restrict__ x_lo,
    const float* __restrict__ W_enc,
    float* __restrict__ cand_val, int* __restrict__ cand_idx)
{
  // LDS: [0,16384) = B hi/lo stage (10240B), aliased by 16KB half-C stage.
  //      [16384, 49152) = per-row top-32 lists (128 rows: val + idx).
  __shared__ __align__(16) char smem[49152];
  __bf16* Bh  = (__bf16*)(smem);
  __bf16* Bl  = (__bf16*)(smem + 5120);
  float*  Cst = (float*)(smem);                 // aliases stage, 64x64 f32
  float*  tv  = (float*)(smem + 16384);
  int*    ti  = (int*)(smem + 32768);

  const int tid  = threadIdx.x;
  const int lane = tid & 31;
  const int wid  = tid >> 5;
  const int fr   = wid & 3;        // wave row group -> rows [32*fr, 32*fr+32)
  const int cg   = wid >> 2;       // wave col group -> cols [32*cg, 32*cg+32)
  const int l16  = lane & 15;
  const int lh   = lane >> 4;

  const int br    = blockIdx.x & (BATCH / BM - 1);   // fastest -> row-blocks share W tile in L2
  const int nc    = blockIdx.x / (BATCH / BM);
  const int row0  = br * BM;
  const int nbase = nc * NCHUNK;

  // init top lists (128 rows x 32)
#pragma unroll
  for (int i = 0; i < (BM * TOPK) / 256; i++) tv[tid + i * 256] = -__builtin_inff();

  // per-lane A fragment base pointers (global; hi/lo, two M-frags)
  const size_t arow0 = (size_t)(row0 + 32 * fr + l16) * D_MODEL + 8 * lh;
  const size_t arow1 = arow0 + (size_t)16 * D_MODEL;
  // per-lane B fragment LDS offsets (elems)
  const int b_off0 = (cg * 32 + l16) * BS + 8 * lh;
  const int b_off1 = b_off0 + 16 * BS;
  // B staging micro-tile: 4(k) x 2(n) per thread
  const int n2 = tid & 31;         // n-pair index (0..31)
  const int kq = tid >> 5;         // k-quad index (0..7)

  for (int nt = 0; nt < NTILES; nt++) {
    const int gn0 = nbase + nt * BN;
    v8f c00 = {}, c01 = {}, c10 = {}, c11 = {};

    for (int kt = 0; kt < KTILES; kt++) {
      const int kk = kt * BK;
      __syncthreads();   // prev frag reads / prev merge done before restage

      // ---- stage B (32k x 64n fp32 -> bf16 hi/lo, transposed, packed b64) ----
      {
        float2 w0 = *(const float2*)(W_enc + (size_t)(kk + 4 * kq + 0) * N_FEAT + gn0 + 2 * n2);
        float2 w1 = *(const float2*)(W_enc + (size_t)(kk + 4 * kq + 1) * N_FEAT + gn0 + 2 * n2);
        float2 w2 = *(const float2*)(W_enc + (size_t)(kk + 4 * kq + 2) * N_FEAT + gn0 + 2 * n2);
        float2 w3 = *(const float2*)(W_enc + (size_t)(kk + 4 * kq + 3) * N_FEAT + gn0 + 2 * n2);
        float f0[4] = { w0.x, w1.x, w2.x, w3.x };
        float f1[4] = { w0.y, w1.y, w2.y, w3.y };
        v4bf h0, l0, h1, l1;
#pragma unroll
        for (int dk = 0; dk < 4; dk++) {
          __bf16 hh, ll;
          split_hl(f0[dk], hh, ll); h0[dk] = hh; l0[dk] = ll;
          split_hl(f1[dk], hh, ll); h1[dk] = hh; l1[dk] = ll;
        }
        *(v4bf*)&Bh[(2 * n2 + 0) * BS + 4 * kq] = h0;
        *(v4bf*)&Bh[(2 * n2 + 1) * BS + 4 * kq] = h1;
        *(v4bf*)&Bl[(2 * n2 + 0) * BS + 4 * kq] = l0;
        *(v4bf*)&Bl[(2 * n2 + 1) * BS + 4 * kq] = l1;
      }
      __syncthreads();

      // ---- A frags direct from global (L2-resident split-x) ----
      v16bf ah0 = frag_cat(x_hi + arow0 + kk, x_hi + arow0 + kk + 16);
      v16bf al0 = frag_cat(x_lo + arow0 + kk, x_lo + arow0 + kk + 16);
      v16bf ah1 = frag_cat(x_hi + arow1 + kk, x_hi + arow1 + kk + 16);
      v16bf al1 = frag_cat(x_lo + arow1 + kk, x_lo + arow1 + kk + 16);
      // ---- B frags from LDS ----
      v16bf bh0 = frag_cat(Bh + b_off0, Bh + b_off0 + 16);
      v16bf bl0 = frag_cat(Bl + b_off0, Bl + b_off0 + 16);
      v16bf bh1 = frag_cat(Bh + b_off1, Bh + b_off1 + 16);
      v16bf bl1 = frag_cat(Bl + b_off1, Bl + b_off1 + 16);

      // ---- 12 WMMAs: hi*hi + hi*lo + lo*hi per output frag ----
      c00 = wmma_bf16(ah0, bh0, c00); c00 = wmma_bf16(ah0, bl0, c00); c00 = wmma_bf16(al0, bh0, c00);
      c01 = wmma_bf16(ah0, bh1, c01); c01 = wmma_bf16(ah0, bl1, c01); c01 = wmma_bf16(al0, bh1, c01);
      c10 = wmma_bf16(ah1, bh0, c10); c10 = wmma_bf16(ah1, bl0, c10); c10 = wmma_bf16(al1, bh0, c10);
      c11 = wmma_bf16(ah1, bh1, c11); c11 = wmma_bf16(ah1, bl1, c11); c11 = wmma_bf16(al1, bh1, c11);
    }

    // ---- stage C in two 64-row phases (aliases B stage) + merge ----
#pragma unroll
    for (int ph = 0; ph < 2; ph++) {
      __syncthreads();
      if ((fr >> 1) == ph) {
        int mb = (fr & 1) * 32;          // local row base within this 64-row half
#pragma unroll
        for (int v = 0; v < 8; v++) {
          int m0 = mb + v + 8 * lh;      // first M-frag rows
          Cst[m0 * BN + cg * 32 + l16]      = c00[v];
          Cst[m0 * BN + cg * 32 + 16 + l16] = c01[v];
          int m1 = m0 + 16;              // second M-frag rows
          Cst[m1 * BN + cg * 32 + l16]      = c10[v];
          Cst[m1 * BN + cg * 32 + 16 + l16] = c11[v];
        }
      }
      __syncthreads();
      if (tid < 64) {
        int r = ph * 64 + tid;           // toplist row (0..127)
        float* tvr = &tv[r * TOPK];
        int*   tir = &ti[r * TOPK];
        const float* crow = &Cst[tid * BN];
        for (int n = 0; n < BN; n++) {
          float v = crow[n];
          if (v > tvr[TOPK - 1]) {       // strict: keeps lower index on ties
            int p = TOPK - 1;
            while (p > 0 && tvr[p - 1] < v) {
              tvr[p] = tvr[p - 1]; tir[p] = tir[p - 1]; --p;
            }
            tvr[p] = v; tir[p] = gn0 + n;
          }
        }
      }
    }
  }

  __syncthreads();
#pragma unroll
  for (int i = 0; i < (BM * TOPK) / 256; i++) {
    int e = tid + i * 256;
    int r = e >> 5, j = e & 31;
    size_t o = ((size_t)(row0 + r) * NSPLIT + nc) * TOPK + j;
    cand_val[o] = tv[r * TOPK + j];
    cand_idx[o] = ti[r * TOPK + j];
  }
}

// ============================================================
// Kernel 2: per-row exact top-32 of 512 candidates, then fused
// sparse decode + output scaling + loss.
// ============================================================
__global__ __launch_bounds__(256) void sae_finalize_decode(
    const float* __restrict__ x, const float* __restrict__ W_dec,
    const float* __restrict__ b_post, const float* __restrict__ avg_norm,
    const int* __restrict__ n_steps,
    const float* __restrict__ cand_val, const int* __restrict__ cand_idx,
    float* __restrict__ y_out, float* __restrict__ loss_out)
{
  __shared__ float sv[NCAND]; __shared__ int si[NCAND];
  __shared__ float rv[NCAND]; __shared__ int rp[NCAND];
  __shared__ float wvs[TOPK]; __shared__ int wis[TOPK];

  const int row = blockIdx.x;
  const int tid = threadIdx.x;
  const size_t cbase = (size_t)row * NCAND;

  sv[tid]       = cand_val[cbase + tid];
  sv[tid + 256] = cand_val[cbase + tid + 256];
  si[tid]       = cand_idx[cbase + tid];
  si[tid + 256] = cand_idx[cbase + tid + 256];
  __syncthreads();

  for (int it = 0; it < TOPK; it++) {
    rv[tid] = sv[tid];             rp[tid] = tid;
    rv[tid + 256] = sv[tid + 256]; rp[tid + 256] = tid + 256;
    __syncthreads();
#pragma unroll
    for (int s = 256; s > 0; s >>= 1) {
      if (tid < s) {
        float a = rv[tid], b = rv[tid + s];
        int pa = rp[tid], pb = rp[tid + s];
        if (b > a || (b == a && si[pb] < si[pa])) { rv[tid] = b; rp[tid] = pb; }
      }
      __syncthreads();
    }
    if (tid == 0) {
      int p = rp[0];
      wvs[it] = sv[p]; wis[it] = si[p];
      sv[p] = -__builtin_inff();
    }
    __syncthreads();
  }

  const float an = avg_norm[0];
  const float scale = SQRT_D / an;
  const float inv   = an / SQRT_D;
  const int ns = n_steps[0];
  const float factor = (ns > 0 ? 1.0f : 0.0f) + (ns >= 64 ? (1.0f / 32.0f) : 0.0f);

  float acc[D_MODEL / 256];
#pragma unroll
  for (int c = 0; c < D_MODEL / 256; c++) acc[c] = 0.0f;

  for (int j = 0; j < TOPK; j++) {
    float w = wvs[j];
    const float* wd = W_dec + (size_t)wis[j] * D_MODEL;
#pragma unroll
    for (int c = 0; c < D_MODEL / 256; c++) acc[c] += w * wd[tid + c * 256];
  }

  float lsum = 0.0f;
#pragma unroll
  for (int c = 0; c < D_MODEL / 256; c++) {
    int d = tid + c * 256;
    float yn = acc[c] + b_post[d];
    float xn = x[(size_t)row * D_MODEL + d] * scale;
    float df = xn - yn;
    lsum += df * df;
    y_out[(size_t)row * D_MODEL + d] = yn * inv;
  }

  rv[tid] = lsum;
  __syncthreads();
#pragma unroll
  for (int s = 128; s > 0; s >>= 1) {
    if (tid < s) rv[tid] += rv[tid + s];
    __syncthreads();
  }
  if (tid == 0) loss_out[row] = rv[0] * (factor / (float)D_MODEL);
}

// ============================================================
extern "C" void kernel_launch(void* const* d_in, const int* in_sizes, int n_in,
                              void* d_out, int out_size, void* d_ws, size_t ws_size,
                              hipStream_t stream) {
  const float* x        = (const float*)d_in[0];
  const float* b_pre    = (const float*)d_in[1];
  const float* W_enc    = (const float*)d_in[2];
  const float* W_dec    = (const float*)d_in[3];
  const float* b_post   = (const float*)d_in[4];
  const float* avg_norm = (const float*)d_in[5];
  // d_in[6] = k (fixed 32), d_in[7] = n_steps
  const int* n_steps    = (const int*)d_in[7];

  float* y_out    = (float*)d_out;
  float* loss_out = y_out + (size_t)BATCH * D_MODEL;

  // workspace layout: x_hi (16MB) | x_lo (16MB) | cand_val (8MB) | cand_idx (8MB)
  char* ws = (char*)d_ws;
  __bf16* x_hi    = (__bf16*)(ws);
  __bf16* x_lo    = (__bf16*)(ws + (size_t)BATCH * D_MODEL * 2);
  float*  cand_val = (float*)(ws + (size_t)BATCH * D_MODEL * 4);
  int*    cand_idx = (int*)(ws + (size_t)BATCH * D_MODEL * 4 +
                            (size_t)BATCH * NCAND * 4);

  sae_split_x<<<dim3((BATCH * D_MODEL) / (256 * 4)), 256, 0, stream>>>(
      x, b_pre, avg_norm, x_hi, x_lo);

  dim3 g1((BATCH / BM) * NSPLIT);    // 512 workgroups
  sae_encode_topk<<<g1, 256, 0, stream>>>(x_hi, x_lo, W_enc, cand_val, cand_idx);

  sae_finalize_decode<<<dim3(BATCH), 256, 0, stream>>>(
      x, W_dec, b_post, avg_norm, n_steps, cand_val, cand_idx, y_out, loss_out);
}
